// _ExpertGroup_57217554317361
// MI455X (gfx1250) — compile-verified, matrix operand order checked by reading, not run on hardware
//
#include <hip/hip_runtime.h>
#include <math.h>
#include <stdint.h>

// ---------------------------------------------------------------------------
// MoE per-token SwiGLU for MI455X (gfx1250).
// Two bf16-WMMA GEMM passes through a bf16 hidden workspace:
//   pass1: hidden = silu(x@Wg^T) * (x@Wu^T)   (per-expert token groups)
//   pass2: out    = hidden @ Wd^T
// M=64 token tiles, N split across 8 waves (16 cols each, no B duplication).
// A tiles double-buffered in LDS via global_load_async_to_lds_b128:
// stage(c+1) issued before compute(c); s_wait_asynccnt 4 exploits in-order
// async completion so the gather latency hides behind WMMA work.
// ---------------------------------------------------------------------------

#define NTOK 8192
#define DMODEL 1024
#define HID 2048
#define NEXP 8
#define TM 64            // tokens per block
#define KB 128           // K chunk staged in LDS per iteration
#define LDS_STRIDE (KB + 8)

typedef __attribute__((ext_vector_type(16))) __bf16          v16bf;
typedef __attribute__((ext_vector_type(8)))  float           v8f;
typedef __attribute__((ext_vector_type(8)))  unsigned short  v8u;
typedef __attribute__((ext_vector_type(16))) unsigned short  v16u;
typedef __attribute__((ext_vector_type(4)))  float           v4f;
typedef __attribute__((ext_vector_type(4)))  unsigned short  v4u;

// round-to-nearest-even fp32 -> bf16
__device__ __forceinline__ unsigned short f2bf(float f) {
    unsigned int u = __builtin_bit_cast(unsigned int, f);
    unsigned int r = u + 0x7FFFu + ((u >> 16) & 1u);
    return (unsigned short)(r >> 16);
}

__device__ __forceinline__ v16bf mkfrag(v8u lo, v8u hi) {
    union { v16u w; v8u h[2]; } u;
    u.h[0] = lo; u.h[1] = hi;
    return __builtin_bit_cast(v16bf, u.w);
}

// CDNA5 async copy: 16B global -> LDS, tracked by ASYNCcnt
__device__ __forceinline__ void async_b128(const unsigned short* src,
                                           unsigned short* ldsdst) {
    unsigned       ldsoff = (unsigned)(uintptr_t)ldsdst;     // low 32b = LDS offset
    unsigned long long ga = (unsigned long long)(uintptr_t)src;
    asm volatile("global_load_async_to_lds_b128 %0, %1, off"
                 :: "v"(ldsoff), "v"(ga) : "memory");
}
__device__ __forceinline__ void wait_async0() {              // all asyncs done
    asm volatile("s_wait_asynccnt 0x0" ::: "memory");
}
__device__ __forceinline__ void wait_async4() {              // all but last stage
    asm volatile("s_wait_asynccnt 0x4" ::: "memory");
}

// Stage a TM x KB bf16 tile (gathered rows) into LDS via async copies.
// 256 threads: thread = (row 0..63, 64B col chunk 0..3) -> 4 x b128 each.
__device__ __forceinline__ void stage_tile(const unsigned short* __restrict__ base,
                                           size_t rowStride, const int* rowsS,
                                           int colBase,
                                           unsigned short (*At)[LDS_STRIDE],
                                           int tid) {
    const int r  = tid & 63;
    const int cc = (tid >> 6) * 32;              // 0,32,64,96 (ushort cols)
    const unsigned short* src = base + (size_t)rowsS[r] * rowStride + colBase + cc;
    unsigned short* dst = &At[r][cc];
    #pragma unroll
    for (int i = 0; i < 4; ++i) async_b128(src + i * 8, dst + i * 8);
}

// ---------------------------------------------------------------------------
// Kernel 1: zero per-expert counters
// ---------------------------------------------------------------------------
__global__ void k_zero(int* cnt) {
    if (threadIdx.x < NEXP) cnt[threadIdx.x] = 0;
}

// ---------------------------------------------------------------------------
// Kernel 2: per-expert token permutation lists
// ---------------------------------------------------------------------------
__global__ void k_perm(const int* __restrict__ eid, int* cnt, int* perm) {
    int n = blockIdx.x * 256 + threadIdx.x;
    if (n < NTOK) {
        int e = eid[n] & (NEXP - 1);
        int s = atomicAdd(&cnt[e], 1);
        perm[e * NTOK + s] = n;
    }
}

// ---------------------------------------------------------------------------
// Kernel 3a: convert weights fp32 -> bf16 (100 MB total, L2-resident)
// ---------------------------------------------------------------------------
__global__ void k_convw(const float* __restrict__ gw, const float* __restrict__ uw,
                        const float* __restrict__ dw,
                        unsigned short* __restrict__ gb,
                        unsigned short* __restrict__ ub,
                        unsigned short* __restrict__ db) {
    size_t i4 = ((size_t)blockIdx.x * 256 + threadIdx.x) * 4;
    if (i4 >= (size_t)NEXP * HID * DMODEL) return;
    v4f g = *(const v4f*)(gw + i4);
    v4f u = *(const v4f*)(uw + i4);
    v4f d = *(const v4f*)(dw + i4);
    v4u go, uo, dn;
    #pragma unroll
    for (int j = 0; j < 4; ++j) { go[j]=f2bf(g[j]); uo[j]=f2bf(u[j]); dn[j]=f2bf(d[j]); }
    *(v4u*)(gb + i4) = go;
    *(v4u*)(ub + i4) = uo;
    *(v4u*)(db + i4) = dn;
}

// ---------------------------------------------------------------------------
// Kernel 3b: convert x fp32 -> bf16 (16 MB)
// ---------------------------------------------------------------------------
__global__ void k_convx(const float* __restrict__ x, unsigned short* __restrict__ xb) {
    size_t i4 = ((size_t)blockIdx.x * 256 + threadIdx.x) * 4;
    if (i4 >= (size_t)NTOK * DMODEL) return;
    v4f v = *(const v4f*)(x + i4);
    v4u o;
    #pragma unroll
    for (int j = 0; j < 4; ++j) o[j] = f2bf(v[j]);
    *(v4u*)(xb + i4) = o;
}

// ---------------------------------------------------------------------------
// Pass 1: hidden = silu(x@Wg^T) * (x@Wu^T), per expert.
// grid = E * (NTOK/TM) * (HID/128); block = 256 (8 waves).
// ---------------------------------------------------------------------------
__global__ __launch_bounds__(256, 1)
void k_pass1(const unsigned short* __restrict__ x_bf,     // [N][D]
             const unsigned short* __restrict__ gate_bf,  // [E][H][D]
             const unsigned short* __restrict__ up_bf,    // [E][H][D]
             const int* __restrict__ cnt, const int* __restrict__ perm,
             unsigned short* __restrict__ hid_bf) {       // [N][H]
    const int MB = NTOK / TM;            // 128
    const int NB = HID / 128;            // 16
    int bid = blockIdx.x;
    const int e  = bid / (MB * NB);
    const int mb = (bid % (MB * NB)) / NB;
    const int nb = bid % NB;

    const int ce = cnt[e];
    if (mb * TM >= ce) return;
    const int valid = (ce - mb * TM < TM) ? (ce - mb * TM) : TM;

    const int tid  = threadIdx.x;
    const int lane = tid & 31;
    const int w    = tid >> 5;
    const int half = lane >> 4;
    const int nIdx = lane & 15;
    const int koffA = half * 8;
    const int h0 = nb * 128 + w * 16;    // this wave's 16 hidden columns

    __shared__ int rowsS[TM];
    __shared__ unsigned short At[2][TM][LDS_STRIDE];   // double-buffered (~35 KB)

    if (tid < TM) {
        int g = mb * TM + tid;
        if (g >= ce) g = ce - 1;
        rowsS[tid] = perm[e * NTOK + g];
    }
    __syncthreads();

    v8f gacc[4], uacc[4];
    #pragma unroll
    for (int mt = 0; mt < 4; ++mt) { gacc[mt] = (v8f)(0.0f); uacc[mt] = (v8f)(0.0f); }

    const unsigned short* gRow = gate_bf + ((size_t)e * HID + h0 + nIdx) * DMODEL;
    const unsigned short* uRow = up_bf   + ((size_t)e * HID + h0 + nIdx) * DMODEL;

    const int NCH = DMODEL / KB;         // 8 chunks
    stage_tile(x_bf, DMODEL, rowsS, 0, At[0], tid);

    for (int c = 0; c < NCH; ++c) {
        const int kb = c * KB;
        if (c + 1 < NCH) {               // stage next chunk, keep it in flight
            stage_tile(x_bf, DMODEL, rowsS, kb + KB, At[(c + 1) & 1], tid);
            __builtin_prefetch(gRow + kb + KB, 0, 1);
            __builtin_prefetch(uRow + kb + KB, 0, 1);
            wait_async4();               // in-order: chunk c resident
        } else {
            wait_async0();
        }
        __syncthreads();

        const unsigned short (*buf)[LDS_STRIDE] = At[c & 1];
        #pragma unroll
        for (int k0 = 0; k0 < KB; k0 += 32) {
            v16bf bg = mkfrag(*(const v8u*)(gRow + kb + k0 + koffA),
                              *(const v8u*)(gRow + kb + k0 + 16 + koffA));
            v16bf bu = mkfrag(*(const v8u*)(uRow + kb + k0 + koffA),
                              *(const v8u*)(uRow + kb + k0 + 16 + koffA));
            #pragma unroll
            for (int mt = 0; mt < 4; ++mt) {
                v16bf a = mkfrag(*(const v8u*)&buf[mt * 16 + nIdx][k0 + koffA],
                                 *(const v8u*)&buf[mt * 16 + nIdx][k0 + 16 + koffA]);
                gacc[mt] = __builtin_amdgcn_wmma_f32_16x16x32_bf16(
                    false, a, false, bg, (short)0, gacc[mt], false, false);
                uacc[mt] = __builtin_amdgcn_wmma_f32_16x16x32_bf16(
                    false, a, false, bu, (short)0, uacc[mt], false, false);
            }
        }
        __syncthreads();                 // buf[c&1] free for stage(c+2)
    }

    // silu(gate)*up -> bf16 hidden, masked tail
    #pragma unroll
    for (int mt = 0; mt < 4; ++mt) {
        #pragma unroll
        for (int j = 0; j < 8; ++j) {
            int m = mt * 16 + j + 8 * half;
            if (m < valid) {
                float gv = gacc[mt][j];
                float hv = (gv / (1.0f + __expf(-gv))) * uacc[mt][j];
                hid_bf[(size_t)rowsS[m] * HID + h0 + nIdx] = f2bf(hv);
            }
        }
    }
}

// ---------------------------------------------------------------------------
// Pass 2: out = hidden @ Wd^T, per expert.
// grid = E * (NTOK/TM) * (DMODEL/128); block = 256 (8 waves).
// ---------------------------------------------------------------------------
__global__ __launch_bounds__(256, 1)
void k_pass2(const unsigned short* __restrict__ hid_bf,   // [N][H]
             const unsigned short* __restrict__ down_bf,  // [E][D][H]
             const int* __restrict__ cnt, const int* __restrict__ perm,
             float* __restrict__ out) {                   // [N][D]
    const int MB = NTOK / TM;            // 128
    const int NB = DMODEL / 128;         // 8
    int bid = blockIdx.x;
    const int e  = bid / (MB * NB);
    const int mb = (bid % (MB * NB)) / NB;
    const int nb = bid % NB;

    const int ce = cnt[e];
    if (mb * TM >= ce) return;
    const int valid = (ce - mb * TM < TM) ? (ce - mb * TM) : TM;

    const int tid  = threadIdx.x;
    const int lane = tid & 31;
    const int w    = tid >> 5;
    const int half = lane >> 4;
    const int nIdx = lane & 15;
    const int koffA = half * 8;
    const int d0 = nb * 128 + w * 16;    // this wave's 16 output columns

    __shared__ int rowsS[TM];
    __shared__ unsigned short At[2][TM][LDS_STRIDE];

    if (tid < TM) {
        int g = mb * TM + tid;
        if (g >= ce) g = ce - 1;
        rowsS[tid] = perm[e * NTOK + g];
    }
    __syncthreads();

    v8f acc[4];
    #pragma unroll
    for (int mt = 0; mt < 4; ++mt) acc[mt] = (v8f)(0.0f);

    const unsigned short* dRow = down_bf + ((size_t)e * DMODEL + d0 + nIdx) * HID;

    const int NCH = HID / KB;            // 16 chunks
    stage_tile(hid_bf, HID, rowsS, 0, At[0], tid);

    for (int c = 0; c < NCH; ++c) {
        const int kb = c * KB;
        if (c + 1 < NCH) {
            stage_tile(hid_bf, HID, rowsS, kb + KB, At[(c + 1) & 1], tid);
            __builtin_prefetch(dRow + kb + KB, 0, 1);
            wait_async4();
        } else {
            wait_async0();
        }
        __syncthreads();

        const unsigned short (*buf)[LDS_STRIDE] = At[c & 1];
        #pragma unroll
        for (int k0 = 0; k0 < KB; k0 += 32) {
            v16bf b = mkfrag(*(const v8u*)(dRow + kb + k0 + koffA),
                             *(const v8u*)(dRow + kb + k0 + 16 + koffA));
            #pragma unroll
            for (int mt = 0; mt < 4; ++mt) {
                v16bf a = mkfrag(*(const v8u*)&buf[mt * 16 + nIdx][k0 + koffA],
                                 *(const v8u*)&buf[mt * 16 + nIdx][k0 + 16 + koffA]);
                acc[mt] = __builtin_amdgcn_wmma_f32_16x16x32_bf16(
                    false, a, false, b, (short)0, acc[mt], false, false);
            }
        }
        __syncthreads();
    }

    #pragma unroll
    for (int mt = 0; mt < 4; ++mt) {
        #pragma unroll
        for (int j = 0; j < 8; ++j) {
            int m = mt * 16 + j + 8 * half;
            if (m < valid)
                out[(size_t)rowsS[m] * DMODEL + d0 + nIdx] = acc[mt][j];
        }
    }
}

// ---------------------------------------------------------------------------
// Host-side launcher
// ---------------------------------------------------------------------------
extern "C" void kernel_launch(void* const* d_in, const int* in_sizes, int n_in,
                              void* d_out, int out_size, void* d_ws, size_t ws_size,
                              hipStream_t stream) {
    const float* x   = (const float*)d_in[0];
    const float* gw  = (const float*)d_in[1];
    const float* uw  = (const float*)d_in[2];
    const float* dw  = (const float*)d_in[3];
    const int*   eid = (const int*)d_in[4];
    float*       out = (float*)d_out;

    const size_t WE = (size_t)NEXP * HID * DMODEL;   // 16,777,216
    const size_t XE = (size_t)NTOK * DMODEL;         //  8,388,608
    const size_t HE = (size_t)NTOK * HID;            // 16,777,216

    unsigned short* gate_bf = (unsigned short*)d_ws;
    unsigned short* up_bf   = gate_bf + WE;
    unsigned short* down_bf = up_bf + WE;
    unsigned short* x_bf    = down_bf + WE;
    unsigned short* hid_bf  = x_bf + XE;
    int* cnt  = (int*)(hid_bf + HE);
    int* perm = cnt + 64;

    k_zero<<<1, 64, 0, stream>>>(cnt);
    k_perm<<<(NTOK + 255) / 256, 256, 0, stream>>>(eid, cnt, perm);
    k_convw<<<(unsigned)((WE / 4 + 255) / 256), 256, 0, stream>>>(
        gw, uw, dw, gate_bf, up_bf, down_bf);
    k_convx<<<(unsigned)((XE / 4 + 255) / 256), 256, 0, stream>>>(x, x_bf);

    k_pass1<<<NEXP * (NTOK / TM) * (HID / 128), 256, 0, stream>>>(
        x_bf, gate_bf, up_bf, cnt, perm, hid_bf);
    k_pass2<<<NEXP * (NTOK / TM) * (DMODEL / 128), 256, 0, stream>>>(
        hid_bf, down_bf, cnt, perm, out);
}
